// DNC_23905787969625
// MI455X (gfx1250) — matrix operand order repaired and checked by reading, not compiled, not run on hardware
//
#include <hip/hip_runtime.h>

// ---------------------------------------------------------------------------
// DNC forward for MI455X (gfx1250, wave32).
// Big per-step GEMMs run on v_wmma_f32_16x16x32_bf16 (f32 accumulate); weights
// are packed to bf16 once per call and stay resident in the 192MB L2.
// The per-batch memory module (sort/cumprod/link/softmax) runs one workgroup
// per batch element with N=128 cells mapped to 128 threads.
// ---------------------------------------------------------------------------

typedef __bf16 bf16_t;
typedef __attribute__((ext_vector_type(16))) __bf16 bf16x16;
typedef __attribute__((ext_vector_type(8)))  __bf16 bf16x8;
typedef __attribute__((ext_vector_type(8)))  float  f32x8;

#define DNC_B   32
#define DNC_T   128
#define DNC_IN  256
#define DNC_H   512
#define DNC_N   128
#define DNC_R   4
#define DNC_W   64
#define DNC_IF  471
#define DNC_IFP 480      // IF padded to a multiple of 16
#define DNC_EPS 1e-6f

__device__ __forceinline__ float sigmoidf_(float x) { return 1.f / (1.f + expf(-x)); }
__device__ __forceinline__ float softplusf_(float x) { return (x > 20.f) ? x : log1pf(expf(x)); }

// ---------------------------------------------------------------------------
// Generic M=32 GEMM:  C(32 x N) = A(32 x K) * B^T   (+ bias)
// B is passed as the weight matrix W (N x K) row-major, i.e. column-major KxN,
// so each lane's WMMA B-fragment is contiguous in memory.
// A can be split at k==ksplit between two sources (x_t lives in its own
// bf16 tensor; last_read/h live in the packed activation buffer).
// Per ISA 7.12.2 (wave32):
//   A 16x32 bf16 : lane(l,half) holds row l, a[0..7]=K half*8+0..7,
//                  a[8..15]=K 16+half*8+0..7
//   B 32x16 bf16 : lane(l,half) holds col l, b[0..7]=K half*8+0..7,
//                  b[8..15]=K 16+half*8+0..7
//   C/D f32      : lane(l,half) acc[j] -> (row half*8+j, col l)
// ---------------------------------------------------------------------------
__global__ __launch_bounds__(64) void gemm_bf16_wmma(
    const bf16_t* __restrict__ A0, int lda0,
    const bf16_t* __restrict__ A1, int lda1, int ksplit,
    const bf16_t* __restrict__ Bw, int ldb,
    const float* __restrict__ bias,
    float* __restrict__ C, long ldc, int ktiles)
{
  const int wave  = threadIdx.x >> 5;   // M-tile (0 or 1)
  const int lane  = threadIdx.x & 31;
  const int kh    = lane >> 4;          // half of the wave
  const int l     = lane & 15;
  const int ntile = blockIdx.x;
  const int arow  = wave * 16 + l;
  const int bcol  = ntile * 16 + l;

  f32x8 acc = {0.f, 0.f, 0.f, 0.f, 0.f, 0.f, 0.f, 0.f};
  for (int kt = 0; kt < ktiles; ++kt) {
    const bf16_t* Ap;
    long lda;
    if (kt * 32 < ksplit) { Ap = A0; lda = lda0; }
    else                  { Ap = A1; lda = lda1; }

    const bf16_t* ap = Ap + (long)arow * lda + kt * 32 + kh * 8;
    bf16x8 alo = *(const bf16x8*)(ap);
    bf16x8 ahi = *(const bf16x8*)(ap + 16);

    const bf16_t* bp = Bw + (long)bcol * ldb + kt * 32 + kh * 8;
    bf16x8 blo = *(const bf16x8*)(bp);
    bf16x8 bhi = *(const bf16x8*)(bp + 16);

    bf16x16 av, bv;
#pragma unroll
    for (int i = 0; i < 8; ++i) {
      av[i] = alo[i]; av[8 + i] = ahi[i];
      bv[i] = blo[i]; bv[8 + i] = bhi[i];
    }
    acc = __builtin_amdgcn_wmma_f32_16x16x32_bf16(
        false, av, false, bv, (short)0, acc, false, false);
  }

  const float bn = bias ? bias[bcol] : 0.f;
#pragma unroll
  for (int j = 0; j < 8; ++j) {
    int row = wave * 16 + kh * 8 + j;
    C[(long)row * ldc + bcol] = acc[j] + bn;
  }
}

// ---------------------------------------------------------------------------
// LSTM pointwise update: gates(32x2048, order i|f|g|o) + c -> c, h(bf16->Abuf)
// ---------------------------------------------------------------------------
__global__ __launch_bounds__(256) void lstm_update(
    const float* __restrict__ gates, float* __restrict__ c,
    bf16_t* __restrict__ Abuf)
{
  int idx = blockIdx.x * 256 + threadIdx.x;
  if (idx >= DNC_B * DNC_H) return;
  int b = idx >> 9;
  int j = idx & 511;
  const float* g = gates + b * 2048;
  float iv = sigmoidf_(g[j]);
  float fv = sigmoidf_(g[512 + j]);
  float gv = tanhf(g[1024 + j]);
  float ov = sigmoidf_(g[1536 + j]);
  float cn = fv * c[idx] + iv * gv;
  c[idx] = cn;
  Abuf[b * 1024 + 512 + j] = (bf16_t)(ov * tanhf(cn));
}

// ---------------------------------------------------------------------------
// Per-batch DNC memory module. One workgroup per batch element, 128 threads,
// thread n == memory cell n.
// ---------------------------------------------------------------------------
__global__ __launch_bounds__(128) void dnc_memory(
    const float* __restrict__ xi,
    float* __restrict__ mem, float* __restrict__ link, float* __restrict__ prec,
    float* __restrict__ read_w, float* __restrict__ write_w,
    float* __restrict__ usage, bf16_t* __restrict__ Abuf)
{
  const int b   = blockIdx.x;
  const int tid = threadIdx.x;
  const int n   = tid;

  __shared__ float xs[DNC_IFP];
  __shared__ float s_rwo[4 * 128];
  __shared__ float s_po[128];
  __shared__ float s_su[128];
  __shared__ int   s_si[128];
  __shared__ float s_cp[128];
  __shared__ float s_alloc[128];
  __shared__ float s_ww[128];
  __shared__ float s_rwn[4 * 128];
  __shared__ float s_red[128];

  const float* xib = xi + b * DNC_IFP;
  float* memb = mem    + (long)b * DNC_N * DNC_W;
  float* lk   = link   + (long)b * DNC_N * DNC_N;
  float* pr   = prec   + b * DNC_N;
  float* rw   = read_w + b * DNC_R * DNC_N;
  float* ww   = write_w + b * DNC_N;
  float* us   = usage  + b * DNC_N;

  // ---- parse + activate interface vector (b_if already fused in GEMM) ----
  for (int idx = tid; idx < DNC_IF; idx += 128) {
    float v = xib[idx];
    float o;
    if      (idx < 256) o = tanhf(v);        // read keys
    else if (idx < 260) o = softplusf_(v);   // read strengths
    else if (idx < 324) o = tanhf(v);        // write key
    else if (idx < 325) o = softplusf_(v);   // write strength
    else if (idx < 389) o = sigmoidf_(v);    // erase
    else if (idx < 453) o = tanhf(v);        // write vector
    else if (idx < 459) o = sigmoidf_(v);    // free gates, alloc gate, write gate
    else                o = v;               // read modes (softmax below)
    xs[idx] = o;
  }
  for (int i = tid; i < 4 * 128; i += 128) s_rwo[i] = rw[i];
  s_po[tid] = pr[tid];
  __syncthreads();
  if (tid < 4) {
    float a0 = xs[459 + tid * 3], a1 = xs[460 + tid * 3], a2 = xs[461 + tid * 3];
    float mx = fmaxf(a0, fmaxf(a1, a2));
    float e0 = expf(a0 - mx), e1 = expf(a1 - mx), e2 = expf(a2 - mx);
    float s = e0 + e1 + e2;
    xs[459 + tid * 3] = e0 / s; xs[460 + tid * 3] = e1 / s; xs[461 + tid * 3] = e2 / s;
  }
  __syncthreads();

  const float* rk    = xs;         // [4][64]
  const float* rstr  = xs + 256;   // [4]
  const float* wkey  = xs + 260;   // [64]
  const float  wstr  = xs[324];
  const float* ers   = xs + 325;   // [64]
  const float* wvec  = xs + 389;   // [64]
  const float* fg    = xs + 453;   // [4]
  const float  ag    = xs[457];
  const float  wgt   = xs[458];
  const float* modes = xs + 459;   // [4][3]

  // ---- retention psi + usage update ----
  float psi = 1.f;
#pragma unroll
  for (int r = 0; r < 4; ++r) psi *= (1.f - fg[r] * s_rwo[r * 128 + n]);
  float uo = us[n], wo = ww[n];
  float u = (uo + wo - uo * wo) * psi;
  us[n] = u;
  s_su[tid] = u; s_si[tid] = tid;
  __syncthreads();

  // ---- stable ascending bitonic argsort of usage (tie-break on index) ----
  for (int k = 2; k <= 128; k <<= 1) {
    for (int j = k >> 1; j > 0; j >>= 1) {
      int ixj = tid ^ j;
      if (ixj > tid) {
        bool up = ((tid & k) == 0);
        float ua = s_su[tid], ub = s_su[ixj];
        int ia = s_si[tid], ib = s_si[ixj];
        bool a_gt_b = (ua > ub) || (ua == ub && ia > ib);
        if (a_gt_b == up) {
          s_su[tid] = ub; s_su[ixj] = ua;
          s_si[tid] = ib; s_si[ixj] = ia;
        }
      }
      __syncthreads();
    }
  }

  // ---- exclusive cumprod of sorted usage -> allocation weighting ----
  s_cp[tid] = s_su[tid];
  __syncthreads();
  for (int off = 1; off < 128; off <<= 1) {
    float v = s_cp[tid];
    if (tid >= off) v *= s_cp[tid - off];
    __syncthreads();
    s_cp[tid] = v;
    __syncthreads();
  }
  float excl = (tid == 0) ? 1.f : s_cp[tid - 1];
  s_alloc[s_si[tid]] = (1.f - s_su[tid]) * excl;
  __syncthreads();

  // ---- write content addressing on OLD memory ----
  float kn2 = 0.f;
#pragma unroll
  for (int w = 0; w < 64; ++w) { float v = wkey[w]; kn2 += v * v; }
  float kinv = 1.f / (sqrtf(kn2) + DNC_EPS);
  float mrow[64];
  float dot = 0.f, m2 = 0.f;
#pragma unroll
  for (int w = 0; w < 64; ++w) {
    float mv = memb[n * 64 + w];
    mrow[w] = mv; dot += wkey[w] * mv; m2 += mv * mv;
  }
  float logit = dot * kinv / (sqrtf(m2) + DNC_EPS) * wstr;
  s_red[tid] = logit; __syncthreads();
  for (int off = 64; off > 0; off >>= 1) { if (tid < off) s_red[tid] = fmaxf(s_red[tid], s_red[tid + off]); __syncthreads(); }
  float mx = s_red[0]; __syncthreads();
  float e = expf(logit - mx);
  s_red[tid] = e; __syncthreads();
  for (int off = 64; off > 0; off >>= 1) { if (tid < off) s_red[tid] += s_red[tid + off]; __syncthreads(); }
  float cwn = e / s_red[0]; __syncthreads();

  // ---- write weighting ----
  float wwn = wgt * (ag * s_alloc[n] + (1.f - ag) * cwn);
  s_ww[n] = wwn; ww[n] = wwn;
  s_red[tid] = wwn; __syncthreads();
  for (int off = 64; off > 0; off >>= 1) { if (tid < off) s_red[tid] += s_red[tid + off]; __syncthreads(); }
  float wsum = s_red[0]; __syncthreads();

  // ---- memory update (erase + add), keep new row in registers ----
#pragma unroll
  for (int w = 0; w < 64; ++w) {
    float mv = mrow[w];
    mv = mv * (1.f - wwn * ers[w]) + wwn * wvec[w];
    mrow[w] = mv;
    memb[n * 64 + w] = mv;
  }

  // ---- temporal link matrix + precedence ----
  for (int m = 0; m < 128; ++m) {
    float lv = lk[n * 128 + m];
    lv = (1.f - wwn - s_ww[m]) * lv + wwn * s_po[m];
    lk[n * 128 + m] = (m == n) ? 0.f : lv;
  }
  __syncthreads();
  pr[tid] = (1.f - wsum) * s_po[tid] + s_ww[tid];

  // ---- forward / backward read weights (uses OLD read weights) ----
  float fw[4] = {0.f, 0.f, 0.f, 0.f};
  float bw[4] = {0.f, 0.f, 0.f, 0.f};
  for (int m = 0; m < 128; ++m) {
    float lnm = lk[n * 128 + m];
    float lmn = lk[m * 128 + n];
#pragma unroll
    for (int r = 0; r < 4; ++r) {
      float rv = s_rwo[r * 128 + m];
      fw[r] += lnm * rv;
      bw[r] += lmn * rv;
    }
  }

  // ---- read content addressing on NEW memory ----
  float m2n = 0.f;
#pragma unroll
  for (int w = 0; w < 64; ++w) m2n += mrow[w] * mrow[w];
  float minv = 1.f / (sqrtf(m2n) + DNC_EPS);
  float crv[4];
#pragma unroll
  for (int r = 0; r < 4; ++r) {
    float k2 = 0.f, d = 0.f;
#pragma unroll
    for (int w = 0; w < 64; ++w) { float kv = rk[r * 64 + w]; k2 += kv * kv; d += kv * mrow[w]; }
    float lg = d / (sqrtf(k2) + DNC_EPS) * minv * rstr[r];
    s_red[tid] = lg; __syncthreads();
    for (int off = 64; off > 0; off >>= 1) { if (tid < off) s_red[tid] = fmaxf(s_red[tid], s_red[tid + off]); __syncthreads(); }
    float mxr = s_red[0]; __syncthreads();
    float er = expf(lg - mxr);
    s_red[tid] = er; __syncthreads();
    for (int off = 64; off > 0; off >>= 1) { if (tid < off) s_red[tid] += s_red[tid + off]; __syncthreads(); }
    crv[r] = er / s_red[0]; __syncthreads();
  }

  // ---- new read weights ----
#pragma unroll
  for (int r = 0; r < 4; ++r) {
    float v = modes[r * 3 + 0] * bw[r] + modes[r * 3 + 1] * crv[r] + modes[r * 3 + 2] * fw[r];
    s_rwn[r * 128 + n] = v;
    rw[r * 128 + n] = v;
  }
  __syncthreads();

  // ---- read vectors -> last_read (bf16) into the activation buffer ----
  for (int o = tid; o < 256; o += 128) {
    int r = o >> 6, w = o & 63;
    float acc = 0.f;
    for (int m = 0; m < 128; ++m) acc += s_rwn[r * 128 + m] * memb[m * 64 + w];
    Abuf[b * 1024 + 256 + r * 64 + w] = (bf16_t)acc;
  }
}

// ---------------------------------------------------------------------------
// One-time (per call) weight / input packing to bf16.
// ---------------------------------------------------------------------------
__global__ void prep_wc(const float* __restrict__ wih, const float* __restrict__ whh,
                        bf16_t* __restrict__ wc)
{
  for (long i = blockIdx.x * (long)blockDim.x + threadIdx.x; i < 2048L * 1024;
       i += (long)gridDim.x * blockDim.x) {
    int nr = (int)(i >> 10), k = (int)(i & 1023);
    float v = (k < 512) ? wih[nr * 512 + k] : whh[nr * 512 + (k - 512)];
    wc[i] = (bf16_t)v;
  }
}

__global__ void prep_wif(const float* __restrict__ wif, bf16_t* __restrict__ o)
{
  for (int i = blockIdx.x * blockDim.x + threadIdx.x; i < DNC_IFP * 512;
       i += gridDim.x * blockDim.x) {
    int nr = i >> 9, k = i & 511;
    o[i] = (bf16_t)((nr < DNC_IF) ? wif[nr * 512 + k] : 0.f);
  }
}

__global__ void prep_wemb(const float* __restrict__ we, bf16_t* __restrict__ o)
{
  // Output K layout matches Abuf cols 256..1023: [last_read(256) | h(512)]
  for (int i = blockIdx.x * blockDim.x + threadIdx.x; i < 256 * 768;
       i += gridDim.x * blockDim.x) {
    int nr = i / 768, k = i % 768;
    float v = (k < 256) ? we[nr * 768 + 512 + k] : we[nr * 768 + (k - 256)];
    o[i] = (bf16_t)v;
  }
}

__global__ void prep_x(const float* __restrict__ x, bf16_t* __restrict__ o)
{
  for (long i = blockIdx.x * (long)blockDim.x + threadIdx.x; i < (long)DNC_B * DNC_T * DNC_IN;
       i += (long)gridDim.x * blockDim.x) {
    o[i] = (bf16_t)x[i];
  }
}

__global__ void prep_bias(const float* __restrict__ b_ih, const float* __restrict__ b_hh,
                          const float* __restrict__ b_if, float* __restrict__ bg,
                          float* __restrict__ bif)
{
  int i = blockIdx.x * blockDim.x + threadIdx.x;
  if (i < 2048) bg[i] = b_ih[i] + b_hh[i];
  if (i < DNC_IFP) bif[i] = (i < DNC_IF) ? b_if[i] : 0.f;
}

// ---------------------------------------------------------------------------
// Host orchestration.
// ---------------------------------------------------------------------------
extern "C" void kernel_launch(void* const* d_in, const int* in_sizes, int n_in,
                              void* d_out, int out_size, void* d_ws, size_t ws_size,
                              hipStream_t stream)
{
  (void)in_sizes; (void)n_in; (void)out_size;

  const float* x     = (const float*)d_in[0];
  const float* W_ih  = (const float*)d_in[1];
  const float* W_hh  = (const float*)d_in[2];
  const float* b_ih  = (const float*)d_in[3];
  const float* b_hh  = (const float*)d_in[4];
  const float* W_if  = (const float*)d_in[5];
  const float* b_if  = (const float*)d_in[6];
  const float* W_emb = (const float*)d_in[7];
  const float* b_emb = (const float*)d_in[8];

  char* base = (char*)d_ws;
  size_t off = 0;
  auto carve = [&](size_t bytes) -> char* {
    char* p = base + off;
    off += (bytes + 255) & ~(size_t)255;
    return p;
  };

  bf16_t* Wc   = (bf16_t*)carve(2048L * 1024 * sizeof(bf16_t));
  bf16_t* Wifc = (bf16_t*)carve((size_t)DNC_IFP * 512 * sizeof(bf16_t));
  bf16_t* We   = (bf16_t*)carve(256L * 768 * sizeof(bf16_t));
  bf16_t* xbf  = (bf16_t*)carve((size_t)DNC_B * DNC_T * DNC_IN * sizeof(bf16_t));
  float*  bg   = (float*)carve(2048 * sizeof(float));
  float*  bifp = (float*)carve(DNC_IFP * sizeof(float));

  char* zstart = base + off;                           // everything below is zeroed
  bf16_t* Abuf  = (bf16_t*)carve((size_t)DNC_B * 1024 * sizeof(bf16_t)); // [x|lr|h]
  float*  gates = (float*)carve((size_t)DNC_B * 2048 * sizeof(float));
  float*  xibuf = (float*)carve((size_t)DNC_B * DNC_IFP * sizeof(float));
  float*  cst   = (float*)carve((size_t)DNC_B * DNC_H * sizeof(float));
  float*  memb  = (float*)carve((size_t)DNC_B * DNC_N * DNC_W * sizeof(float));
  float*  lnk   = (float*)carve((size_t)DNC_B * DNC_N * DNC_N * sizeof(float));
  float*  prc   = (float*)carve((size_t)DNC_B * DNC_N * sizeof(float));
  float*  rdw   = (float*)carve((size_t)DNC_B * DNC_R * DNC_N * sizeof(float));
  float*  wrw   = (float*)carve((size_t)DNC_B * DNC_N * sizeof(float));
  float*  usg   = (float*)carve((size_t)DNC_B * DNC_N * sizeof(float));
  size_t zbytes = (size_t)((base + off) - zstart);

  if (off > ws_size) return;  // workspace too small; bail deterministically

  // ---- one-time packing + state zeroing (all capture-legal) ----
  hipMemsetAsync(zstart, 0, zbytes, stream);
  prep_wc  <<<2048, 256, 0, stream>>>(W_ih, W_hh, Wc);
  prep_wif <<<960,  256, 0, stream>>>(W_if, Wifc);
  prep_wemb<<<768,  256, 0, stream>>>(W_emb, We);
  prep_x   <<<2048, 256, 0, stream>>>(x, xbf);
  prep_bias<<<8,    256, 0, stream>>>(b_ih, b_hh, b_if, bg, bifp);

  float* out = (float*)d_out;
  const long ldx = (long)DNC_T * DNC_IN;               // 32768

  for (int t = 0; t < DNC_T; ++t) {
    // gates(32x2048) = [x_t | last_read | h] @ [W_ih|W_hh]^T + (b_ih+b_hh)
    gemm_bf16_wmma<<<128, 64, 0, stream>>>(
        xbf + (long)t * DNC_IN, (int)ldx,   // k < 256 : x_t
        Abuf, 1024, 256,                    // k >= 256: last_read | h
        Wc, 1024, bg, gates, 2048, 32);

    // LSTM pointwise: c update, h -> Abuf[512..1023] (bf16)
    lstm_update<<<64, 256, 0, stream>>>(gates, cst, Abuf);

    // xi(32x480) = h @ W_if^T + b_if (padded)
    gemm_bf16_wmma<<<30, 64, 0, stream>>>(
        Abuf + 512, 1024, Abuf + 512, 1024, 0,
        Wifc, 512, bifp, xibuf, DNC_IFP, 16);

    // memory module: updates mem/link/prec/read_w/write_w/usage,
    // writes last_read -> Abuf[256..511] (bf16)
    dnc_memory<<<DNC_B, 128, 0, stream>>>(xibuf, memb, lnk, prc, rdw, wrw, usg, Abuf);

    // out(32x256) = [h | last_read] @ W_emb^T + b_emb  (K layout matches We)
    gemm_bf16_wmma<<<16, 64, 0, stream>>>(
        Abuf + 256, 1024, Abuf + 256, 1024, 0,
        We, 768, b_emb, out + (long)t * 256, (long)DNC_T * 256, 24);
  }
}
